// HarmonicIntegration_48155173323216
// MI455X (gfx1250) — matrix operand order repaired and checked by reading, not compiled, not run on hardware
//
#include <hip/hip_runtime.h>
#include <math.h>

// ---------------------------------------------------------------------------
// Problem dims (compile-time constants from the reference)
// ---------------------------------------------------------------------------
#define BB 8
#define CC 64
#define TT 512
#define FF 256
#define NN 128
#define HD 64      // H*D
#define FTILE 128  // freq tile for LDS-resident phases

typedef __bf16 bf16;
typedef __attribute__((ext_vector_type(16))) __bf16 v16bf;
typedef __attribute__((ext_vector_type(8)))  __bf16 v8bf;
typedef __attribute__((ext_vector_type(8)))  float  v8f;

// Tensor-DMA descriptor vector types (per probe-verified builtin signature)
typedef __attribute__((ext_vector_type(4))) unsigned int u32x4;
typedef __attribute__((ext_vector_type(4))) int          i32x4;
typedef __attribute__((ext_vector_type(8))) int          i32x8;

// ---------------------------------------------------------------------------
// WMMA fragment helpers (CDNA5 16x16x32 bf16, wave32; A/B use contiguous k order (16 per lane half), C/D per ISA §7.12.2)
// ---------------------------------------------------------------------------
__device__ __forceinline__ v8f wmma_bf16(v16bf a, v16bf b, v8f c) {
    return __builtin_amdgcn_wmma_f32_16x16x32_bf16(
        /*neg_a=*/false, a, /*neg_b=*/false, b,
        /*c_mod=*/(short)0, c, /*reuse_a=*/false, /*reuse_b=*/false);
}

// A (16x32, row-major src with leading dim ld): lane = (m, half);
// lanes 0-15 hold K 0..15, lanes 16-31 hold K 16..31.
__device__ __forceinline__ v16bf load_a_frag(const bf16* base, int ld, int m0, int k0) {
    const int lane = threadIdx.x & 31;
    const int m = lane & 15, hlf = lane >> 4;
    const bf16* p = base + (size_t)(m0 + m) * ld + k0 + hlf * 16;
    v8bf lo = *reinterpret_cast<const v8bf*>(p);
    v8bf hi = *reinterpret_cast<const v8bf*>(p + 8);
    v16bf a;
#pragma unroll
    for (int i = 0; i < 8; ++i) { a[i] = lo[i]; a[8 + i] = hi[i]; }
    return a;
}

// B (32x16): element (k, n) lives at base + n*ld + k (column-contiguous).
// Lane n holds 16 consecutive K values starting at k0 + 16*(lane>=16).
__device__ __forceinline__ v16bf load_b_frag(const bf16* base, int ld, int k0, int n0) {
    const int lane = threadIdx.x & 31;
    const int n = lane & 15, hlf = lane >> 4;
    const bf16* p = base + (size_t)(n0 + n) * ld + k0 + hlf * 16;
    v8bf lo = *reinterpret_cast<const v8bf*>(p);
    v8bf hi = *reinterpret_cast<const v8bf*>(p + 8);
    v16bf b;
#pragma unroll
    for (int i = 0; i < 8; ++i) { b[i] = lo[i]; b[8 + i] = hi[i]; }
    return b;
}

// Guarded B fragment: columns outside [nlo, nhi] contribute zeros (conv halo).
__device__ __forceinline__ v16bf load_b_frag_guard(const bf16* base, int ld, int k0,
                                                   int n0, int nlo, int nhi) {
    const int lane = threadIdx.x & 31;
    const int n = n0 + (lane & 15), hlf = lane >> 4;
    v16bf b = {};
    if (n >= nlo && n <= nhi) {
        const bf16* p = base + (size_t)n * ld + k0 + hlf * 16;
        v8bf lo = *reinterpret_cast<const v8bf*>(p);
        v8bf hi = *reinterpret_cast<const v8bf*>(p + 8);
#pragma unroll
        for (int i = 0; i < 8; ++i) { b[i] = lo[i]; b[8 + i] = hi[i]; }
    }
    return b;
}

// C/D layout: vgpr j holds (m = m0 + 8*half + j, n = n0 + lane&15).
__device__ __forceinline__ v8f acc_bias(const float* bias, int m0) {
    const int lane = threadIdx.x & 31;
    const int hlf = lane >> 4;
    v8f c;
#pragma unroll
    for (int j = 0; j < 8; ++j) c[j] = bias[m0 + hlf * 8 + j];
    return c;
}

__device__ __forceinline__ void store_d_rowmajor_bf16(bf16* dst, int ld, int m0, int n0, v8f c) {
    const int lane = threadIdx.x & 31;
    const int n = lane & 15, hlf = lane >> 4;
#pragma unroll
    for (int j = 0; j < 8; ++j)
        dst[(size_t)(m0 + hlf * 8 + j) * ld + n0 + n] = (bf16)c[j];
}

// Store D transposed: dst[n][m] — contiguous 16B per lane, feeds next GEMM's B.
__device__ __forceinline__ void store_d_transposed_bf16(bf16* dst, int ld, int m0, int n0, v8f c) {
    const int lane = threadIdx.x & 31;
    const int n = lane & 15, hlf = lane >> 4;
    bf16* p = dst + (size_t)(n0 + n) * ld + m0 + hlf * 8;
#pragma unroll
    for (int j = 0; j < 8; ++j) p[j] = (bf16)c[j];
}

__device__ __forceinline__ void store_d_rowmajor_f32(float* dst, int ld, int m0, int n0,
                                                     v8f c, float scale) {
    const int lane = threadIdx.x & 31;
    const int n = lane & 15, hlf = lane >> 4;
#pragma unroll
    for (int j = 0; j < 8; ++j)
        dst[(size_t)(m0 + hlf * 8 + j) * ld + n0 + n] = c[j] * scale;
}

// mod = harm2 * V, stored transposed [f][hd] as B for the output GEMM.
__device__ __forceinline__ void store_mod_bf16(bf16* mt, const bf16* vt, int ld,
                                               int m0, int n0, v8f c) {
    const int lane = threadIdx.x & 31;
    const int n = lane & 15, hlf = lane >> 4;
    const size_t col = (size_t)(n0 + n) * ld;
#pragma unroll
    for (int j = 0; j < 8; ++j) {
        int m = m0 + hlf * 8 + j;
        float v = (float)vt[col + m];
        mt[col + m] = (bf16)(c[j] * v);
    }
}

__device__ __forceinline__ void store_out_f32(float* out, int b, int t, int m0, int nf0, v8f c) {
    const int lane = threadIdx.x & 31;
    const int n = lane & 15, hlf = lane >> 4;
#pragma unroll
    for (int j = 0; j < 8; ++j) {
        int m = m0 + hlf * 8 + j;
        out[(((size_t)b * CC + m) * TT + t) * FF + nf0 + n] = c[j];
    }
}

// ---------------------------------------------------------------------------
// Tensor Data Mover: DMA a contiguous 16 KB global block into LDS.
// D# per ISA §8.3/§8.4: group0 = {count=1, lds_addr, global_addr[56:0], type=2},
// group1 = {data_size=8B, tensor_dim0 = tile_dim0 = 2048, stride = 2048}.
// Groups 2/3 zero (<=2D tensor). One wave issues; tracked by TENSORcnt.
// ---------------------------------------------------------------------------
__device__ __forceinline__ void tdm_load_16kb(const bf16* gsrc, unsigned lds_byte) {
    const unsigned long long ga = (unsigned long long)(uintptr_t)gsrc;
    u32x4 g0;
    g0[0] = 1u;                                                 // count=1, user-mode
    g0[1] = lds_byte;                                           // lds_addr[63:32]
    g0[2] = (unsigned)(ga & 0xffffffffull);                     // global_addr[95:64]
    g0[3] = (unsigned)((ga >> 32) & 0x1ffffffull) | 0x80000000u;// addr hi | type=2
    i32x8 g1;
    g1[0] = (int)(3u << 16);       // data_size = 8 bytes, workgroup_mask = 0
    g1[1] = (int)(2048u << 16);    // tensor_dim0[15:0] (2048 x 8B = 16 KB)
    g1[2] = (int)(1u << 16);       // tensor_dim0 hi = 0, tensor_dim1 = 1
    g1[3] = (int)(2048u << 16);    // tile_dim0 = 2048
    g1[4] = 1;                     // tile_dim1 = 1, tile_dim2 = 0
    g1[5] = 2048;                  // tensor_dim0_stride[31:0]
    g1[6] = 0;
    g1[7] = 0;
    i32x4 gz = {};
#if defined(__clang_major__) && (__clang_major__ >= 23)
    i32x8 gz8 = {};
    __builtin_amdgcn_tensor_load_to_lds(g0, g1, gz, gz, gz8, 0);
#else
    __builtin_amdgcn_tensor_load_to_lds(g0, g1, gz, gz, 0);
#endif
}

// ---------------------------------------------------------------------------
// Prep kernels: fp32 -> bf16 staging with WMMA-friendly layouts
// ---------------------------------------------------------------------------
__global__ void prep_weights(const float* __restrict__ Wk, const float* __restrict__ Wv,
                             const float* __restrict__ Wh, const float* __restrict__ Wo,
                             bf16* __restrict__ Wkb, bf16* __restrict__ Wvb,
                             bf16* __restrict__ Whb, bf16* __restrict__ Wob) {
    int i = blockIdx.x * blockDim.x + threadIdx.x;
    if (i < HD * CC) {
        Wkb[i] = (bf16)Wk[i];
        Wvb[i] = (bf16)Wv[i];
        Wob[i] = (bf16)Wo[i];
    }
    if (i < 3 * HD * HD) {  // Wh OIHW (64,64,1,3) -> Whb[tap][o][i] dense taps
        int tap = i / (HD * HD);
        int rem = i - tap * (HD * HD);
        int o = rem >> 6, ii = rem & 63;
        Whb[i] = (bf16)Wh[(o * HD + ii) * 3 + tap];
    }
}

// X (B,C,T,F) f32 -> Xt[b][t][f][c] bf16 (column-contiguous B for K/V GEMMs)
__global__ void prep_x(const float* __restrict__ X, bf16* __restrict__ Xt, long total) {
    for (long i = (long)blockIdx.x * blockDim.x + threadIdx.x; i < total;
         i += (long)gridDim.x * blockDim.x) {
        int c = (int)(i & 63);
        long r = i >> 6;
        int f = (int)(r & 255); r >>= 8;
        int t = (int)(r & 511); r >>= 9;
        int b = (int)r;
        Xt[i] = (bf16)X[(((long)b * CC + c) * TT + t) * FF + f];
    }
}

// Q (B,N,F) -> Qbf[b][n][f] and Qt[b][f][n] (bf16)
__global__ void prep_q(const float* __restrict__ Q, bf16* __restrict__ Qbf,
                       bf16* __restrict__ Qt) {
    long i = (long)blockIdx.x * blockDim.x + threadIdx.x;
    if (i >= (long)BB * NN * FF) return;
    int f = (int)(i & 255);
    long r = i >> 8;
    int n = (int)(r & 127);
    int b = (int)(r >> 7);
    float q = Q[i];
    Qbf[i] = (bf16)q;
    Qt[((long)b * FF + f) * NN + n] = (bf16)q;
}

// ---------------------------------------------------------------------------
// Fused main kernel: one workgroup (8 wave32) per (b, t) slice.
// GEMM chain: K/V proj -> scores -> softmax -> harm -> 3-tap conv -> V-mod -> out
// X tiles staged LDS-side via the Tensor Data Mover.
// ---------------------------------------------------------------------------
__global__ __launch_bounds__(256)
void harmint_main(const bf16* __restrict__ Xt,   // [B][T][F][C]
                  const bf16* __restrict__ Qbf,  // [B][N][F]
                  const bf16* __restrict__ Qtb_g,// [B][F][N]
                  const bf16* __restrict__ Wkb,  // [HD][C]
                  const bf16* __restrict__ Wvb,  // [HD][C]
                  const bf16* __restrict__ Whb,  // [3][HD][HD]
                  const bf16* __restrict__ Wob,  // [C][HD]
                  const float* __restrict__ bk, const float* __restrict__ bv,
                  const float* __restrict__ bh, const float* __restrict__ bo,
                  float* __restrict__ out) {
    const int bt = blockIdx.x;
    const int b = bt >> 9;        // / TT
    const int t = bt & (TT - 1);
    const int wave = threadIdx.x >> 5;

    // 64 KB LDS, live-range aliased.
    __shared__ alignas(16) unsigned char smA[32768]; // K-tile + X stage | f32 scores | harm^T
    __shared__ alignas(16) unsigned char smB[16384]; // softmax weights | X stage | mod tile
    __shared__ alignas(16) unsigned char smC[16384]; // V tile (transposed)

    bf16*  KsT = (bf16*)smA;             // [HD][FTILE] row-major (A of scores GEMM)
    bf16*  Xs  = (bf16*)(smA + 16384);   // [FTILE][CC] TDM-staged X tile (phase 1)
    float* Sc  = (float*)smA;            // [HD][NN] f32 logits
    bf16*  Ht  = (bf16*)smA;             // [FF][HD]  harm transposed (B of conv GEMMs)
    bf16*  Ws  = (bf16*)smB;             // [HD][NN]  softmax weights (A of harm GEMM)
    bf16*  XsB = (bf16*)smB;             // [FTILE][CC] TDM-staged X tile (phase 5a)
    bf16*  Mt  = (bf16*)smB;             // [FTILE][HD] mod transposed (B of out GEMM)
    bf16*  Vt  = (bf16*)smC;             // [FTILE][HD] V transposed

    const bf16* Xbt = Xt + ((size_t)b * TT + t) * FF * CC;  // [F][C]
    const bf16* Qb  = Qbf + (size_t)b * NN * FF;            // [N][F]
    const bf16* Qtb = Qtb_g + (size_t)b * FF * NN;          // [F][N]

    // Pull this batch's Q tiles toward the caches (L2-resident, reused by 512 WGs)
    if (threadIdx.x < NN) {
        __builtin_prefetch(Qb + (size_t)threadIdx.x * FF, 0, 2);
        __builtin_prefetch(Qtb + (size_t)threadIdx.x * NN * 2, 0, 2);
    }

    // ---- Phase 1+2: K projection + scores accumulation over F tiles --------
    v8f sacc[4];
#pragma unroll
    for (int i = 0; i < 4; ++i) sacc[i] = (v8f){};

#pragma unroll
    for (int ft = 0; ft < 2; ++ft) {
        const int f0 = ft * FTILE;
        // TDM: DMA this f-tile of X (16 KB, contiguous [f][c]) into LDS.
        if (wave == 0) {
            tdm_load_16kb(Xbt + (size_t)f0 * CC, (unsigned)(uintptr_t)Xs);
            __builtin_amdgcn_s_wait_tensorcnt(0);
        }
        __syncthreads();
        // K tile: 64 x 128 = 4x8 tiles of 16x16, 4 tiles per wave.
#pragma unroll
        for (int i = 0; i < 4; ++i) {
            const int tid = wave * 4 + i;
            const int m0 = (tid >> 3) * 16, n0 = (tid & 7) * 16;
            v8f c = acc_bias(bk, m0);
#pragma unroll
            for (int k = 0; k < CC; k += 32) {
                v16bf a = load_a_frag(Wkb, CC, m0, k);
                v16bf bm = load_b_frag(Xs, CC, k, n0);      // LDS-staged X
                c = wmma_bf16(a, bm, c);
            }
            store_d_rowmajor_bf16(KsT, FTILE, m0, n0, c);
        }
        __syncthreads();
        // scores += K_tile (64xFTILE) * Q^T_tile (FTILExNN)
#pragma unroll
        for (int i = 0; i < 4; ++i) {
            const int tid = wave * 4 + i;
            const int m0 = (tid >> 3) * 16, n0 = (tid & 7) * 16;
#pragma unroll
            for (int k = 0; k < FTILE; k += 32) {
                v16bf a = load_a_frag(KsT, FTILE, m0, k);
                v16bf bq = load_b_frag(Qb, FF, f0 + k, n0);
                sacc[i] = wmma_bf16(a, bq, sacc[i]);
            }
        }
        __syncthreads();
    }

    // ---- Phase 3: scaled logits -> LDS, row softmax, bf16 weights ----------
    const float scale = 0.25f;  // D^-0.5, D=16
#pragma unroll
    for (int i = 0; i < 4; ++i) {
        const int tid = wave * 4 + i;
        store_d_rowmajor_f32(Sc, NN, (tid >> 3) * 16, (tid & 7) * 16, sacc[i], scale);
    }
    __syncthreads();
    if (threadIdx.x < HD) {
        float* row = Sc + (size_t)threadIdx.x * NN;
        float mx = -1e30f;
        for (int n = 0; n < NN; ++n) mx = fmaxf(mx, row[n]);
        float s = 0.f;
        for (int n = 0; n < NN; ++n) { float e = __expf(row[n] - mx); row[n] = e; s += e; }
        const float inv = 1.f / s;
        bf16* wr = Ws + (size_t)threadIdx.x * NN;
        for (int n = 0; n < NN; ++n) wr[n] = (bf16)(row[n] * inv);
    }
    __syncthreads();

    // ---- Phase 4: harm = w (64x128) * Q (128x256) -> Ht[f][hd] -------------
#pragma unroll
    for (int i = 0; i < 8; ++i) {
        const int tid = wave * 8 + i;           // 64 tiles: 4 x 16
        const int m0 = (tid >> 4) * 16, n0 = (tid & 15) * 16;
        v8f c = (v8f){};
#pragma unroll
        for (int k = 0; k < NN; k += 32) {
            v16bf a = load_a_frag(Ws, NN, m0, k);
            v16bf bq = load_b_frag(Qtb, NN, k, n0);
            c = wmma_bf16(a, bq, c);
        }
        store_d_transposed_bf16(Ht, HD, m0, n0, c);
    }
    __syncthreads();

    // ---- Phase 5: per F tile — V proj, conv+mod, output GEMM ---------------
#pragma unroll
    for (int ft = 0; ft < 2; ++ft) {
        const int f0 = ft * FTILE;
        // TDM: re-stage this f-tile of X into smB (Ws / previous Mt are dead).
        if (wave == 0) {
            tdm_load_16kb(Xbt + (size_t)f0 * CC, (unsigned)(uintptr_t)XsB);
            __builtin_amdgcn_s_wait_tensorcnt(0);
        }
        __syncthreads();
        // (a) V tile -> Vt[f_local][hd]
#pragma unroll
        for (int i = 0; i < 4; ++i) {
            const int tid = wave * 4 + i;
            const int m0 = (tid >> 3) * 16, n0 = (tid & 7) * 16;
            v8f c = acc_bias(bv, m0);
#pragma unroll
            for (int k = 0; k < CC; k += 32) {
                v16bf a = load_a_frag(Wvb, CC, m0, k);
                v16bf bm = load_b_frag(XsB, CC, k, n0);     // LDS-staged X
                c = wmma_bf16(a, bm, c);
            }
            store_d_transposed_bf16(Vt, HD, m0, n0, c);
        }
        __syncthreads();
        // (b) 3-tap freq conv over Ht (halo zero-guarded), then mod = V * harm2
        //     (Mt overwrites the X stage — V projection is complete, barrier above)
#pragma unroll
        for (int i = 0; i < 4; ++i) {
            const int tid = wave * 4 + i;
            const int m0 = (tid >> 3) * 16, n0 = (tid & 7) * 16;
            v8f c = acc_bias(bh, m0);
#pragma unroll
            for (int tap = 0; tap < 3; ++tap) {
                const bf16* A = Whb + tap * HD * HD;
                const int col0 = f0 + n0 + tap - 1;  // global freq column base
#pragma unroll
                for (int k = 0; k < HD; k += 32) {
                    v16bf a = load_a_frag(A, HD, m0, k);
                    v16bf bm = load_b_frag_guard(Ht, HD, k, col0, 0, FF - 1);
                    c = wmma_bf16(a, bm, c);
                }
            }
            store_mod_bf16(Mt, Vt, HD, m0, n0, c);
        }
        __syncthreads();
        // (c) out = Wo (64x64) * mod (64xFTILE) + bo -> global fp32
#pragma unroll
        for (int i = 0; i < 4; ++i) {
            const int tid = wave * 4 + i;
            const int m0 = (tid >> 3) * 16, n0 = (tid & 7) * 16;
            v8f c = acc_bias(bo, m0);
#pragma unroll
            for (int k = 0; k < HD; k += 32) {
                v16bf a = load_a_frag(Wob, HD, m0, k);
                v16bf bm = load_b_frag(Mt, HD, k, n0);
                c = wmma_bf16(a, bm, c);
            }
            store_out_f32(out, b, t, m0, f0 + n0, c);
        }
        __syncthreads();
    }
}

// ---------------------------------------------------------------------------
// Host launcher
// ---------------------------------------------------------------------------
extern "C" void kernel_launch(void* const* d_in, const int* in_sizes, int n_in,
                              void* d_out, int out_size, void* d_ws, size_t ws_size,
                              hipStream_t stream) {
    const float* X  = (const float*)d_in[0];
    const float* Q  = (const float*)d_in[1];
    const float* Wk = (const float*)d_in[2];
    const float* bk = (const float*)d_in[3];
    const float* Wv = (const float*)d_in[4];
    const float* bv = (const float*)d_in[5];
    const float* Wh = (const float*)d_in[6];
    const float* bh = (const float*)d_in[7];
    const float* Wo = (const float*)d_in[8];
    const float* bo = (const float*)d_in[9];
    float* out = (float*)d_out;

    char* ws = (char*)d_ws;
    size_t off = 0;
    bf16* Xt  = (bf16*)(ws + off); off += (size_t)BB * TT * FF * CC * sizeof(bf16);
    bf16* Qbf = (bf16*)(ws + off); off += (size_t)BB * NN * FF * sizeof(bf16);
    bf16* Qt  = (bf16*)(ws + off); off += (size_t)BB * NN * FF * sizeof(bf16);
    bf16* Wkb = (bf16*)(ws + off); off += (size_t)HD * CC * sizeof(bf16);
    bf16* Wvb = (bf16*)(ws + off); off += (size_t)HD * CC * sizeof(bf16);
    bf16* Whb = (bf16*)(ws + off); off += (size_t)3 * HD * HD * sizeof(bf16);
    bf16* Wob = (bf16*)(ws + off); off += (size_t)HD * CC * sizeof(bf16);

    prep_weights<<<48, 256, 0, stream>>>(Wk, Wv, Wh, Wo, Wkb, Wvb, Whb, Wob);

    const long totalX = (long)BB * CC * TT * FF;
    prep_x<<<4096, 256, 0, stream>>>(X, Xt, totalX);

    const int qElems = BB * NN * FF;
    prep_q<<<(qElems + 255) / 256, 256, 0, stream>>>(Q, Qbf, Qt);

    harmint_main<<<BB * TT, 256, 0, stream>>>(Xt, Qbf, Qt, Wkb, Wvb, Whb, Wob,
                                              bk, bv, bh, bo, out);
}
